// GMMEncoder_13615046328719
// MI455X (gfx1250) — matrix operand order repaired and checked by reading, not compiled, not run on hardware
//
#include <hip/hip_runtime.h>
#include <hip/hip_bf16.h>

// ---------------- problem constants (from reference) ----------------
#define BS    4
#define T_    32
#define NN_   128
#define F_IN  64
#define HID   64      // per-head channels C (all layers)
#define HEADS 4
#define NC_   32
#define LD_   64
#define RH_   256
#define E_    2048
#define G_    (BS * T_)            // 128 graphs
#define N_    (G_ * NN_)           // 16384 nodes
#define ET_   (G_ * E_ + N_)       // 278528 edges incl. self loops
#define HC1   (HEADS * HID)        // 256
#define NEG_SLOPE 0.2f

typedef __attribute__((ext_vector_type(16))) _Float16 v16h;
typedef __attribute__((ext_vector_type(8)))  _Float16 v8h;
typedef __attribute__((ext_vector_type(4)))  _Float16 v4h;
typedef __attribute__((ext_vector_type(8)))  float    v8f;

// ---------------- device helpers ----------------
__device__ __forceinline__ void edge_sd(const int* __restrict__ ei, long long t,
                                        int& s, int& d) {
  const long long GE = (long long)G_ * E_;
  if (t < GE) {
    int g = (int)(t / E_);
    int e = (int)(t - (long long)g * E_);
    s = ei[e]      + g * NN_;
    d = ei[E_ + e] + g * NN_;
  } else {
    int n = (int)(t - GE);
    s = n; d = n;
  }
}

__device__ __forceinline__ void atomicMaxF(float* addr, float val) {
  int* ai = (int*)addr;
  int old = *ai;
  while (__int_as_float(old) < val) {
    int prev = atomicCAS(ai, old, __float_as_int(val));
    if (prev == old) break;
    old = prev;
  }
}

__device__ __forceinline__ float sigmoidf(float x) {
  return 1.0f / (1.0f + __expf(-x));
}

// ---------------- WMMA GEMM: C[M,Nn] = A[M,K] @ B[K,Nn], f32 io, f16 math ----
// Block = 256 threads (8 waves) computes a 128x64 C tile.
// Wave w owns rows [w*16, w*16+16) x all 64 cols -> 4 f32 accumulators,
// 4 v_wmma_f32_16x16x32_f16 per k-step with one shared A fragment.
// Per k-step the block stages A(128x32) and B^T(64x32) in LDS as f16 via
// coalesced global_load_b128 + v_cvt_pk_f16_f32; fragments are then two
// contiguous ds_load_b128 per lane (ISA 16x32-A / 32x16-B f16 layouts).
__global__ void __launch_bounds__(256)
wmma_gemm_f16(const float* __restrict__ A, const float* __restrict__ B,
              float* __restrict__ C, int M, int Nn, int K) {
  __shared__ _Float16 sA[128][32];   // [row][k]   8 KB
  __shared__ _Float16 sBt[64][32];   // [col][k]   4 KB (transposed)

  const int tilesN = Nn >> 6;                    // 64-col blocks
  const int bm = blockIdx.x / tilesN;
  const int bn = blockIdx.x % tilesN;
  const int m0 = bm << 7;                        // block row base (128)
  const int n0 = bn << 6;                        // block col base (64)

  const int tid  = threadIdx.x;
  const int lane = tid & 31;
  const int wave = tid >> 5;
  const int lo   = lane & 15;
  const int hi   = lane >> 4;
  const int rm   = wave << 4;                    // wave's row base in tile

  v8f acc0 = {0.f,0.f,0.f,0.f,0.f,0.f,0.f,0.f};
  v8f acc1 = acc0, acc2 = acc0, acc3 = acc0;

  for (int k0 = 0; k0 < K; k0 += 32) {
    // ---- stage A: 128 rows x 32 k (1024 float4 loads, 4 per thread) ----
#pragma unroll
    for (int i = 0; i < 4; ++i) {
      int q = tid + (i << 8);                    // 0..1023
      int r = q >> 3;                            // 0..127
      int c = (q & 7) << 2;                      // 0,4,..,28
      const float4 v = *(const float4*)(A + (size_t)(m0 + r) * K + k0 + c);
      if (k0 + 32 < K) __builtin_prefetch(A + (size_t)(m0 + r) * K + k0 + 32 + c, 0, 0);
      v4h h4 = { (_Float16)v.x, (_Float16)v.y, (_Float16)v.z, (_Float16)v.w };
      *(v4h*)&sA[r][c] = h4;
    }
    // ---- stage B transposed: 32 k x 64 cols (512 float4 loads, 2/thread) ----
#pragma unroll
    for (int i = 0; i < 2; ++i) {
      int q = tid + (i << 8);                    // 0..511
      int kk = q >> 4;                           // 0..31
      int c  = (q & 15) << 2;                    // 0,4,..,60
      const float4 v = *(const float4*)(B + (size_t)(k0 + kk) * Nn + n0 + c);
      sBt[c + 0][kk] = (_Float16)v.x;
      sBt[c + 1][kk] = (_Float16)v.y;
      sBt[c + 2][kk] = (_Float16)v.z;
      sBt[c + 3][kk] = (_Float16)v.w;
    }
    __syncthreads();

    // ---- A fragment (shared by the 4 N-tiles) ----
    v8h a_lo = *(const v8h*)&sA[rm + lo][8 * hi];
    v8h a_hi = *(const v8h*)&sA[rm + lo][16 + 8 * hi];
    v16h a = __builtin_shufflevector(a_lo, a_hi,
                                     0,1,2,3,4,5,6,7,8,9,10,11,12,13,14,15);
    // ---- 4 B fragments + 4 WMMAs ----
#pragma unroll
    for (int nt = 0; nt < 4; ++nt) {
      const int col = (nt << 4) + lo;
      v8h b_lo = *(const v8h*)&sBt[col][8 * hi];
      v8h b_hi = *(const v8h*)&sBt[col][16 + 8 * hi];
      v16h b = __builtin_shufflevector(b_lo, b_hi,
                                       0,1,2,3,4,5,6,7,8,9,10,11,12,13,14,15);
      if      (nt == 0) acc0 = __builtin_amdgcn_wmma_f32_16x16x32_f16(false, a, false, b, (short)0, acc0, false, false);
      else if (nt == 1) acc1 = __builtin_amdgcn_wmma_f32_16x16x32_f16(false, a, false, b, (short)0, acc1, false, false);
      else if (nt == 2) acc2 = __builtin_amdgcn_wmma_f32_16x16x32_f16(false, a, false, b, (short)0, acc2, false, false);
      else              acc3 = __builtin_amdgcn_wmma_f32_16x16x32_f16(false, a, false, b, (short)0, acc3, false, false);
    }
    __syncthreads();
  }

  // ---- store: C/D layout -> row = r + 8*hi, col = lane&15 ----
#pragma unroll
  for (int r = 0; r < 8; ++r) {
    const size_t row = (size_t)(m0 + rm + r + 8 * hi) * Nn + n0 + lo;
    C[row +  0] = acc0[r];
    C[row + 16] = acc1[r];
    C[row + 32] = acc2[r];
    C[row + 48] = acc3[r];
  }
}

// ---------------- GAT support kernels ----------------
__global__ void fill_kernel(float* __restrict__ p, float v, long long n) {
  long long i = (long long)blockIdx.x * blockDim.x + threadIdx.x;
  if (i < n) p[i] = v;
}

__global__ void attn_coef_kernel(const float* __restrict__ H,
                                 const float* __restrict__ as_,
                                 const float* __restrict__ ad_,
                                 float* __restrict__ a_src,
                                 float* __restrict__ a_dst, int Hh) {
  long long idx = (long long)blockIdx.x * blockDim.x + threadIdx.x;
  if (idx >= (long long)N_ * Hh) return;
  int n = (int)(idx / Hh), h = (int)(idx % Hh);
  const float* hp = H + (size_t)n * Hh * HID + h * HID;
  float s1 = 0.f, s2 = 0.f;
#pragma unroll 8
  for (int c = 0; c < HID; ++c) {
    float v = hp[c];
    s1 += v * as_[h * HID + c];
    s2 += v * ad_[h * HID + c];
  }
  a_src[idx] = s1;
  a_dst[idx] = s2;
}

__global__ void edge_max_kernel(const int* __restrict__ ei,
                                const float* __restrict__ a_src,
                                const float* __restrict__ a_dst,
                                float* __restrict__ m, int Hh) {
  long long idx = (long long)blockIdx.x * blockDim.x + threadIdx.x;
  if (idx >= (long long)ET_ * Hh) return;
  long long t = idx / Hh;
  int h = (int)(idx % Hh);
  int s, d; edge_sd(ei, t, s, d);
  float ev = a_src[(size_t)s * Hh + h] + a_dst[(size_t)d * Hh + h];
  ev = ev >= 0.f ? ev : NEG_SLOPE * ev;
  atomicMaxF(&m[(size_t)d * Hh + h], ev);
}

__global__ void edge_exp_kernel(const int* __restrict__ ei,
                                const float* __restrict__ a_src,
                                const float* __restrict__ a_dst,
                                const float* __restrict__ m,
                                float* __restrict__ exb,
                                float* __restrict__ den, int Hh) {
  long long idx = (long long)blockIdx.x * blockDim.x + threadIdx.x;
  if (idx >= (long long)ET_ * Hh) return;
  long long t = idx / Hh;
  int h = (int)(idx % Hh);
  int s, d; edge_sd(ei, t, s, d);
  float ev = a_src[(size_t)s * Hh + h] + a_dst[(size_t)d * Hh + h];
  ev = ev >= 0.f ? ev : NEG_SLOPE * ev;
  float ex = expf(ev - m[(size_t)d * Hh + h]);
  exb[idx] = ex;
  atomicAdd(&den[(size_t)d * Hh + h], ex);
}

__global__ void edge_agg_kernel(const int* __restrict__ ei,
                                const float* __restrict__ H,
                                const float* __restrict__ exb,
                                const float* __restrict__ den,
                                float* __restrict__ out, int Hh) {
  long long idx = (long long)blockIdx.x * blockDim.x + threadIdx.x;
  const int HC = Hh * HID;
  if (idx >= (long long)ET_ * HC) return;
  long long t = idx / HC;
  int c = (int)(idx % HC);
  int h = c / HID;
  int s, d; edge_sd(ei, t, s, d);
  float alpha = exb[t * Hh + h] / (den[(size_t)d * Hh + h] + 1e-16f);
  atomicAdd(&out[(size_t)d * HC + c], alpha * H[(size_t)s * HC + c]);
}

__global__ void bias_relu_kernel(float* __restrict__ x,
                                 const float* __restrict__ b,
                                 int HC, long long n) {
  long long i = (long long)blockIdx.x * blockDim.x + threadIdx.x;
  if (i < n) x[i] = fmaxf(x[i] + b[(int)(i % HC)], 0.f);
}

// ---------------- pooling ----------------
__global__ void pool_kernel(const float* __restrict__ h3,
                            float* __restrict__ ge) {
  int idx = blockIdx.x * blockDim.x + threadIdx.x;  // G_*HID = 8192
  if (idx >= G_ * HID) return;
  int g = idx / HID, c = idx % HID;
  float s = 0.f;
  for (int n = 0; n < NN_; ++n)
    s += h3[(size_t)(g * NN_ + n) * HID + c];
  ge[idx] = s;
}

// ---------------- biLSTM: block 0 = forward, block 1 = backward ----------------
__global__ void __launch_bounds__(1024)
lstm_kernel(const float* __restrict__ ge,  // [G_][HID], g = b*T + t
            const float* __restrict__ Wih_f, const float* __restrict__ Whh_f,
            const float* __restrict__ bih_f, const float* __restrict__ bhh_f,
            const float* __restrict__ Wih_b, const float* __restrict__ Whh_b,
            const float* __restrict__ bih_b, const float* __restrict__ bhh_b,
            float* __restrict__ h_out_f, float* __restrict__ h_out_b) {
  const int dir = blockIdx.x;
  const float* Wi = dir ? Wih_b : Wih_f;
  const float* Wh = dir ? Whh_b : Whh_f;
  const float* bi = dir ? bih_b : bih_f;
  const float* bh = dir ? bhh_b : bhh_f;
  float* hout     = dir ? h_out_b : h_out_f;

  __shared__ float sh[BS][RH_];
  __shared__ float sc[BS][RH_];
  __shared__ float sg[BS][4 * RH_];
  const int tid = threadIdx.x;

  for (int i = tid; i < BS * RH_; i += 1024) {
    (&sh[0][0])[i] = 0.f;
    (&sc[0][0])[i] = 0.f;
  }
  __syncthreads();

  for (int step = 0; step < T_; ++step) {
    const int t = dir ? (T_ - 1 - step) : step;
    const int j = tid;  // gate index, all 4 batch rows
    float acc0, acc1, acc2, acc3;
    {
      float bias = bi[j] + bh[j];
      acc0 = bias; acc1 = bias; acc2 = bias; acc3 = bias;
    }
    for (int k = 0; k < HID; ++k) {
      float w = Wi[(size_t)j * HID + k];
      acc0 += w * ge[((size_t)0 * T_ + t) * HID + k];
      acc1 += w * ge[((size_t)1 * T_ + t) * HID + k];
      acc2 += w * ge[((size_t)2 * T_ + t) * HID + k];
      acc3 += w * ge[((size_t)3 * T_ + t) * HID + k];
    }
    for (int k = 0; k < RH_; ++k) {
      float w = Wh[(size_t)j * RH_ + k];
      acc0 += w * sh[0][k];
      acc1 += w * sh[1][k];
      acc2 += w * sh[2][k];
      acc3 += w * sh[3][k];
    }
    sg[0][j] = acc0; sg[1][j] = acc1; sg[2][j] = acc2; sg[3][j] = acc3;
    __syncthreads();
    const int bb = tid >> 8, jj = tid & 255;
    float iv = sg[bb][jj];
    float fv = sg[bb][RH_ + jj];
    float gv = sg[bb][2 * RH_ + jj];
    float ov = sg[bb][3 * RH_ + jj];
    float cc = sigmoidf(fv) * sc[bb][jj] + sigmoidf(iv) * tanhf(gv);
    sc[bb][jj] = cc;
    sh[bb][jj] = sigmoidf(ov) * tanhf(cc);
    __syncthreads();
  }
  if (tid < BS * RH_) hout[tid] = (&sh[0][0])[tid];
}

// ---------------- output heads ----------------
__global__ void heads_kernel(const float* __restrict__ hf,
                             const float* __restrict__ hb,
                             const float* __restrict__ Wmu,
                             const float* __restrict__ bmu,
                             const float* __restrict__ Wlv,
                             const float* __restrict__ blv,
                             const float* __restrict__ Wpi,
                             const float* __restrict__ bpi,
                             float* __restrict__ out) {
  const int NOUT = BS * NC_ * LD_;  // 8192
  int idx = blockIdx.x * blockDim.x + threadIdx.x;
  if (idx < NOUT) {  // mu
    int b = idx / (NC_ * LD_), i = idx % (NC_ * LD_);
    float s = bmu[i];
    const float* w = Wmu + (size_t)i * (2 * RH_);
    for (int k = 0; k < RH_; ++k) s += hf[b * RH_ + k] * w[k];
    for (int k = 0; k < RH_; ++k) s += hb[b * RH_ + k] * w[RH_ + k];
    out[idx] = s;
  } else if (idx < 2 * NOUT) {  // logvar
    int q = idx - NOUT;
    int b = q / (NC_ * LD_), i = q % (NC_ * LD_);
    float s = blv[i];
    const float* w = Wlv + (size_t)i * (2 * RH_);
    for (int k = 0; k < RH_; ++k) s += hf[b * RH_ + k] * w[k];
    for (int k = 0; k < RH_; ++k) s += hb[b * RH_ + k] * w[RH_ + k];
    out[idx] = s;
  } else if (idx < 2 * NOUT + BS * NC_) {  // pi logits
    int q = idx - 2 * NOUT;
    int b = q / NC_, j = q % NC_;
    float s = bpi[j];
    const float* w = Wpi + (size_t)j * (2 * RH_);
    for (int k = 0; k < RH_; ++k) s += hf[b * RH_ + k] * w[k];
    for (int k = 0; k < RH_; ++k) s += hb[b * RH_ + k] * w[RH_ + k];
    out[idx] = s;
  }
}

// ---------------- host driver ----------------
extern "C" void kernel_launch(void* const* d_in, const int* in_sizes, int n_in,
                              void* d_out, int out_size, void* d_ws, size_t ws_size,
                              hipStream_t stream) {
  (void)in_sizes; (void)n_in; (void)out_size; (void)ws_size;
  const float* x     = (const float*)d_in[0];
  const int*   ei    = (const int*)  d_in[1];
  const float* W1    = (const float*)d_in[2];
  const float* as1   = (const float*)d_in[3];
  const float* ad1   = (const float*)d_in[4];
  const float* b1    = (const float*)d_in[5];
  const float* W2    = (const float*)d_in[6];
  const float* as2   = (const float*)d_in[7];
  const float* ad2   = (const float*)d_in[8];
  const float* b2    = (const float*)d_in[9];
  const float* W3    = (const float*)d_in[10];
  const float* as3   = (const float*)d_in[11];
  const float* ad3   = (const float*)d_in[12];
  const float* b3    = (const float*)d_in[13];
  const float* Wih_f = (const float*)d_in[14];
  const float* Whh_f = (const float*)d_in[15];
  const float* bih_f = (const float*)d_in[16];
  const float* bhh_f = (const float*)d_in[17];
  const float* Wih_b = (const float*)d_in[18];
  const float* Whh_b = (const float*)d_in[19];
  const float* bih_b = (const float*)d_in[20];
  const float* bhh_b = (const float*)d_in[21];
  const float* Wmu   = (const float*)d_in[22];
  const float* bmu   = (const float*)d_in[23];
  const float* Wlv   = (const float*)d_in[24];
  const float* blv   = (const float*)d_in[25];
  const float* Wpi   = (const float*)d_in[26];
  const float* bpi   = (const float*)d_in[27];
  float* out = (float*)d_out;

  // workspace layout (floats)
  float* ws    = (float*)d_ws;
  float* Hbuf  = ws;                                  // [N_, 256]
  float* F0    = Hbuf  + (size_t)N_ * HC1;            // [N_, 256]
  float* F1    = F0    + (size_t)N_ * HC1;            // [N_, 256]
  float* a_src = F1    + (size_t)N_ * HC1;            // [N_, HEADS]
  float* a_dst = a_src + (size_t)N_ * HEADS;
  float* mbuf  = a_dst + (size_t)N_ * HEADS;
  float* den   = mbuf  + (size_t)N_ * HEADS;
  float* exb   = den   + (size_t)N_ * HEADS;          // [ET_, HEADS]
  float* ge    = exb   + (size_t)ET_ * HEADS;         // [G_, HID]
  float* hf    = ge    + (size_t)G_ * HID;            // [BS, RH_]
  float* hb    = hf    + (size_t)BS * RH_;

  auto gb = [](long long n) { return (unsigned)((n + 255) / 256); };

  auto run_gat = [&](const float* in, const float* W, const float* as_,
                     const float* ad_, const float* b_, int K, int Hh,
                     float* Hout, float* Oout) {
    const int Nn = Hh * HID;
    const int blocks = (N_ / 128) * (Nn / 64);
    wmma_gemm_f16<<<blocks, 256, 0, stream>>>(in, W, Hout, N_, Nn, K);
    const long long nh = (long long)N_ * Hh;
    attn_coef_kernel<<<gb(nh), 256, 0, stream>>>(Hout, as_, ad_, a_src, a_dst, Hh);
    fill_kernel<<<gb(nh), 256, 0, stream>>>(mbuf, -1e30f, nh);
    fill_kernel<<<gb(nh), 256, 0, stream>>>(den, 0.f, nh);
    const long long eth = (long long)ET_ * Hh;
    edge_max_kernel<<<gb(eth), 256, 0, stream>>>(ei, a_src, a_dst, mbuf, Hh);
    edge_exp_kernel<<<gb(eth), 256, 0, stream>>>(ei, a_src, a_dst, mbuf, exb, den, Hh);
    const long long nchan = (long long)N_ * Nn;
    fill_kernel<<<gb(nchan), 256, 0, stream>>>(Oout, 0.f, nchan);
    const long long etc = (long long)ET_ * Nn;
    edge_agg_kernel<<<gb(etc), 256, 0, stream>>>(ei, Hout, exb, den, Oout, Hh);
    bias_relu_kernel<<<gb(nchan), 256, 0, stream>>>(Oout, b_, Nn, nchan);
  };

  run_gat(x,  W1, as1, ad1, b1, F_IN, HEADS, Hbuf, F0);  // layer 1
  run_gat(F0, W2, as2, ad2, b2, HC1,  HEADS, Hbuf, F1);  // layer 2
  run_gat(F1, W3, as3, ad3, b3, HC1,  1,     Hbuf, F0);  // layer 3 -> F0 [N_,64]

  pool_kernel<<<gb((long long)G_ * HID), 256, 0, stream>>>(F0, ge);

  lstm_kernel<<<2, 1024, 0, stream>>>(ge, Wih_f, Whh_f, bih_f, bhh_f,
                                      Wih_b, Whh_b, bih_b, bhh_b, hf, hb);

  heads_kernel<<<gb(2LL * BS * NC_ * LD_ + BS * NC_), 256, 0, stream>>>(
      hf, hb, Wmu, bmu, Wlv, blv, Wpi, bpi, out);
}
